// CausalSelfAttention_38079180046347
// MI455X (gfx1250) — compile-verified, matrix operand order checked by reading, not run on hardware
//
#include <hip/hip_runtime.h>
#include <hip/hip_bf16.h>

// ---------------------------------------------------------------------------
// Causal self-attention forward, bf16 WMMA pipeline for gfx1250 (MI455X).
// Stages: cvt/transpose -> QKV GEMM -> flash attention -> out-proj GEMM.
// Round 2: software-pipelined (double-buffered) GEMM loops, 32x64 wave tiles,
// early V-fragment issue in attention, exp2-domain softmax.
// ---------------------------------------------------------------------------

typedef __bf16 bf16_t;
typedef __attribute__((ext_vector_type(16))) __bf16 v16bf;
typedef __attribute__((ext_vector_type(8)))  __bf16 v8bf;
typedef __attribute__((ext_vector_type(8)))  float  v8f;

static constexpr int Bn  = 4;
static constexpr int Tn  = 2048;
static constexpr int Dn  = 1024;
static constexpr int Hn  = 16;
static constexpr int DHn = 64;

// --------------------------- WMMA helpers ----------------------------------

__device__ __forceinline__ v8f wmma_bf16(v16bf a, v16bf b, v8f c) {
  // D(f32 16x16) = A(bf16 16x32) * B(bf16 32x16) + C
  return __builtin_amdgcn_wmma_f32_16x16x32_bf16(
      /*neg_a=*/false, a, /*neg_b=*/false, b,
      /*c_mod=*/(short)0, c, /*reuse_a=*/false, /*reuse_b=*/false);
}

// Load one A/B fragment for lane (16 bf16 values) from a row pointer.
// Per ISA 7.12.2 (16-bit A 16x32): lane's half selects K chunks
//   elems [0..7]  -> K = kbase + half*8 + j
//   elems [8..15] -> K = kbase + 16 + half*8 + (j-8)
// i.e. two contiguous 16-byte chunks.
__device__ __forceinline__ v16bf load_frag(const bf16_t* __restrict__ rowptr,
                                           int kbase, int half) {
  v8bf lo = *(const v8bf*)(rowptr + kbase + half * 8);
  v8bf hi = *(const v8bf*)(rowptr + kbase + 16 + half * 8);
  v16bf r;
#pragma unroll
  for (int i = 0; i < 8; ++i) { r[i] = lo[i]; r[8 + i] = hi[i]; }
  return r;
}

// --------------------------- conversion kernels ----------------------------

__global__ void __launch_bounds__(256) cvt_bf16_kernel(
    const float* __restrict__ in, bf16_t* __restrict__ out, int n) {
  int i = blockIdx.x * 256 + threadIdx.x;
  if (i < n) out[i] = (bf16_t)in[i];
}

// out[c][r] = bf16(in[r][c])   (in: rows x cols row-major)
__global__ void __launch_bounds__(256) transpose_bf16_kernel(
    const float* __restrict__ in, bf16_t* __restrict__ out,
    int rows, int cols) {
  int i = blockIdx.x * 256 + threadIdx.x;
  if (i < rows * cols) {
    int r = i / cols, c = i % cols;
    out[(size_t)c * rows + r] = (bf16_t)in[i];
  }
}

// --------------------------- QKV projection GEMM ---------------------------
// qkv = xbf @ W_qkv + b_qkv ; scatter into Q,K (row-major [B,H,T,64]) and
// Vt (transposed [B,H,64,T]).  One wave computes a 32(M) x 64(N) tile with a
// double-buffered K loop (8 WMMAs per 32-K step, loads one step ahead).

__global__ void __launch_bounds__(256) qkv_gemm_kernel(
    const bf16_t* __restrict__ xbf, const bf16_t* __restrict__ WT,
    const float* __restrict__ bias,
    bf16_t* __restrict__ Qb, bf16_t* __restrict__ Kb,
    bf16_t* __restrict__ Vt) {
  const int lane = threadIdx.x & 31;
  const int wave = threadIdx.x >> 5;
  const int task = blockIdx.x * 8 + wave;
  const int NSTRIP = (3 * Dn) / 64;          // 48
  const int mt = task / NSTRIP;              // 0..255  (B*T/32 strips)
  const int ns = task % NSTRIP;
  const int t0 = mt * 32;                    // row in [0, B*T)
  const int n0 = ns * 64;                    // col in [0, 3*D)
  const int half = lane >> 4;
  const int l16  = lane & 15;

  const bf16_t* arow[2];
#pragma unroll
  for (int mi = 0; mi < 2; ++mi)
    arow[mi] = xbf + (size_t)(t0 + 16 * mi + l16) * Dn;
  const bf16_t* brow[4];
#pragma unroll
  for (int i = 0; i < 4; ++i) brow[i] = WT + (size_t)(n0 + 16 * i + l16) * Dn;

  v8f acc[2][4] = {};
  v16bf a0[2], b0[4], a1[2], b1[4];

  // prologue: k = 0 into buffer 0
#pragma unroll
  for (int mi = 0; mi < 2; ++mi) a0[mi] = load_frag(arow[mi], 0, half);
#pragma unroll
  for (int i = 0; i < 4; ++i) b0[i] = load_frag(brow[i], 0, half);

  for (int k0 = 0; k0 < Dn; k0 += 64) {
    // prefetch k0+32 into buffer 1 (always in range: k0 <= Dn-64)
#pragma unroll
    for (int mi = 0; mi < 2; ++mi) a1[mi] = load_frag(arow[mi], k0 + 32, half);
#pragma unroll
    for (int i = 0; i < 4; ++i) b1[i] = load_frag(brow[i], k0 + 32, half);
    // consume buffer 0
#pragma unroll
    for (int mi = 0; mi < 2; ++mi)
#pragma unroll
      for (int i = 0; i < 4; ++i)
        acc[mi][i] = wmma_bf16(a0[mi], b0[i], acc[mi][i]);
    // prefetch k0+64 into buffer 0
    if (k0 + 64 < Dn) {
#pragma unroll
      for (int mi = 0; mi < 2; ++mi)
        a0[mi] = load_frag(arow[mi], k0 + 64, half);
#pragma unroll
      for (int i = 0; i < 4; ++i) b0[i] = load_frag(brow[i], k0 + 64, half);
    }
    // consume buffer 1
#pragma unroll
    for (int mi = 0; mi < 2; ++mi)
#pragma unroll
      for (int i = 0; i < 4; ++i)
        acc[mi][i] = wmma_bf16(a1[mi], b1[i], acc[mi][i]);
  }

  const int region = n0 >> 10;               // 0=Q 1=K 2=V
  const int n0r = n0 & 1023;
  const int h = n0r >> 6;

#pragma unroll
  for (int mi = 0; mi < 2; ++mi) {
    const int trow = t0 + 16 * mi;
    const int bidx  = trow / Tn;
    const int trow0 = trow % Tn;
#pragma unroll
    for (int i = 0; i < 4; ++i) {
      const int dh = 16 * i + l16;
      const float bv = bias[n0 + 16 * i + l16];
#pragma unroll
      for (int r = 0; r < 8; ++r) {
        const int m = r + 8 * half;          // C layout: M = r + 8*(lane/16)
        const int t = trow0 + m;
        const bf16_t o = (bf16_t)(acc[mi][i][r] + bv);
        if (region == 0)
          Qb[(((size_t)bidx * Hn + h) * Tn + t) * DHn + dh] = o;
        else if (region == 1)
          Kb[(((size_t)bidx * Hn + h) * Tn + t) * DHn + dh] = o;
        else
          Vt[(((size_t)bidx * Hn + h) * DHn + dh) * Tn + t] = o;
      }
    }
  }
}

// --------------------------- flash attention -------------------------------
// One wave owns one 16-query tile of one (b,h). Streams 32 keys per step:
// V fragments issued first (stay in flight through softmax), S = Q@K^T
// (4 K-frag loads then 4 WMMAs), causal mask, exp2-domain online softmax,
// P routed through LDS into A-fragment layout, O += P@V (4 WMMAs).

#define LDSCOL 40  // 32 keys + 8 pad: keeps 16B alignment, spreads banks

__global__ void __launch_bounds__(256) attn_kernel(
    const bf16_t* __restrict__ Qb, const bf16_t* __restrict__ Kb,
    const bf16_t* __restrict__ Vt, bf16_t* __restrict__ Ob) {
  __shared__ bf16_t lds[8][16][LDSCOL];

  const int lane = threadIdx.x & 31;
  const int wave = threadIdx.x >> 5;
  const int task = blockIdx.x * 8 + wave;    // 8192 tasks
  const int qt = task & 127;                 // T/16 = 128
  const int bh = task >> 7;                  // b*H + h
  const int b  = bh >> 4;
  const int h  = bh & 15;
  const int q0 = qt * 16;
  const int half = lane >> 4;
  const int l16  = lane & 15;

  // Q fragment rows (lane m = l16), preloaded for both dh-halves.
  const bf16_t* qrow = Qb + (((size_t)bh) * Tn + q0 + l16) * DHn;
  const v16bf qa0 = load_frag(qrow, 0, half);
  const v16bf qa1 = load_frag(qrow, 32, half);

  const bf16_t* krow_base = Kb + ((size_t)bh) * Tn * DHn;
  const bf16_t* vtrow[4];
#pragma unroll
  for (int i = 0; i < 4; ++i)
    vtrow[i] = Vt + (((size_t)bh) * DHn + 16 * i + l16) * Tn;

  v8f acc[4] = {};
  float mrow[8], lrow[8];
#pragma unroll
  for (int r = 0; r < 8; ++r) { mrow[r] = -1e30f; lrow[r] = 0.f; }

  // scores kept in scaled-log2 domain: s2 = S * (1/sqrt(dh)) * log2(e)
  const float scale2 = 0.125f * 1.44269504088896340736f;

  for (int j0 = 0; j0 <= q0; j0 += 32) {
    // ---- issue V fragment loads first; they stay in flight until P@V ----
    v16bf bv[4];
#pragma unroll
    for (int i = 0; i < 4; ++i) bv[i] = load_frag(vtrow[i], j0, half);

    // ---- K fragments for both 16-key tiles, then score WMMAs ----
    v16bf kb[2][2];
#pragma unroll
    for (int st = 0; st < 2; ++st) {
      const bf16_t* krow = krow_base + (size_t)(j0 + 16 * st + l16) * DHn;
      kb[st][0] = load_frag(krow, 0, half);
      kb[st][1] = load_frag(krow, 32, half);
    }
    v8f s[2];
#pragma unroll
    for (int st = 0; st < 2; ++st) {
      v8f c = {};
      c = wmma_bf16(qa0, kb[st][0], c);
      c = wmma_bf16(qa1, kb[st][1], c);
      s[st] = c;
    }

    // ---- scale (log2 domain) + causal mask ----
#pragma unroll
    for (int st = 0; st < 2; ++st) {
      const int key = j0 + 16 * st + l16;
#pragma unroll
      for (int r = 0; r < 8; ++r) {
        const int q = q0 + r + 8 * half;
        const float v = s[st][r] * scale2;
        s[st][r] = (key <= q) ? v : -1e30f;
      }
    }
    // ---- block row-max (reduce across 16 lanes of each half) ----
    float alpha[8];
#pragma unroll
    for (int r = 0; r < 8; ++r) {
      float v = fmaxf(s[0][r], s[1][r]);
#pragma unroll
      for (int off = 1; off < 16; off <<= 1)
        v = fmaxf(v, __shfl_xor(v, off, 32));
      const float mn = fmaxf(mrow[r], v);
      alpha[r] = exp2f(mrow[r] - mn);
      mrow[r] = mn;
    }
    // ---- P = exp2(s2 - m), row sums, stash P into LDS (C layout) ----
    float rs[8];
#pragma unroll
    for (int r = 0; r < 8; ++r) rs[r] = 0.f;
#pragma unroll
    for (int st = 0; st < 2; ++st) {
#pragma unroll
      for (int r = 0; r < 8; ++r) {
        const float p = exp2f(s[st][r] - mrow[r]);
        rs[r] += p;
        lds[wave][r + 8 * half][16 * st + l16] = (bf16_t)p;
      }
    }
#pragma unroll
    for (int r = 0; r < 8; ++r) {
      float v = rs[r];
#pragma unroll
      for (int off = 1; off < 16; off <<= 1)
        v += __shfl_xor(v, off, 32);
      lrow[r] = lrow[r] * alpha[r] + v;
#pragma unroll
      for (int i = 0; i < 4; ++i) acc[i][r] *= alpha[r];
    }
    // ---- reload P as A fragment (per-wave LDS, compiler orders DS ops) ----
    v8bf plo = *(const v8bf*)&lds[wave][l16][half * 8];
    v8bf phi = *(const v8bf*)&lds[wave][l16][16 + half * 8];
    v16bf pa;
#pragma unroll
    for (int i = 0; i < 8; ++i) { pa[i] = plo[i]; pa[8 + i] = phi[i]; }
    // ---- O += P @ V  (V fragments were issued at loop top) ----
#pragma unroll
    for (int i = 0; i < 4; ++i) acc[i] = wmma_bf16(pa, bv[i], acc[i]);
  }

  // ---- normalize and store O in [B,T,D] (bf16) ----
#pragma unroll
  for (int r = 0; r < 8; ++r) lrow[r] = 1.0f / lrow[r];
#pragma unroll
  for (int i = 0; i < 4; ++i) {
#pragma unroll
    for (int r = 0; r < 8; ++r) {
      const int m = r + 8 * half;
      const bf16_t o = (bf16_t)(acc[i][r] * lrow[r]);
      Ob[((size_t)b * Tn + q0 + m) * Dn + h * DHn + 16 * i + l16] = o;
    }
  }
}

// --------------------------- output projection GEMM ------------------------
// out = Ob @ W_out + b_out  (fp32 result). One wave = 32x64 tile,
// double-buffered K loop.

__global__ void __launch_bounds__(256) oproj_kernel(
    const bf16_t* __restrict__ Ob, const bf16_t* __restrict__ WoT,
    const float* __restrict__ bias, float* __restrict__ out) {
  const int lane = threadIdx.x & 31;
  const int wave = threadIdx.x >> 5;
  const int task = blockIdx.x * 8 + wave;
  const int NSTRIP = Dn / 64;                // 16
  const int mt = task / NSTRIP;              // 0..255
  const int ns = task % NSTRIP;
  const int t0 = mt * 32;
  const int n0 = ns * 64;
  const int half = lane >> 4;
  const int l16  = lane & 15;

  const bf16_t* arow[2];
#pragma unroll
  for (int mi = 0; mi < 2; ++mi)
    arow[mi] = Ob + (size_t)(t0 + 16 * mi + l16) * Dn;
  const bf16_t* brow[4];
#pragma unroll
  for (int i = 0; i < 4; ++i) brow[i] = WoT + (size_t)(n0 + 16 * i + l16) * Dn;

  v8f acc[2][4] = {};
  v16bf a0[2], b0[4], a1[2], b1[4];

#pragma unroll
  for (int mi = 0; mi < 2; ++mi) a0[mi] = load_frag(arow[mi], 0, half);
#pragma unroll
  for (int i = 0; i < 4; ++i) b0[i] = load_frag(brow[i], 0, half);

  for (int k0 = 0; k0 < Dn; k0 += 64) {
#pragma unroll
    for (int mi = 0; mi < 2; ++mi) a1[mi] = load_frag(arow[mi], k0 + 32, half);
#pragma unroll
    for (int i = 0; i < 4; ++i) b1[i] = load_frag(brow[i], k0 + 32, half);
#pragma unroll
    for (int mi = 0; mi < 2; ++mi)
#pragma unroll
      for (int i = 0; i < 4; ++i)
        acc[mi][i] = wmma_bf16(a0[mi], b0[i], acc[mi][i]);
    if (k0 + 64 < Dn) {
#pragma unroll
      for (int mi = 0; mi < 2; ++mi)
        a0[mi] = load_frag(arow[mi], k0 + 64, half);
#pragma unroll
      for (int i = 0; i < 4; ++i) b0[i] = load_frag(brow[i], k0 + 64, half);
    }
#pragma unroll
    for (int mi = 0; mi < 2; ++mi)
#pragma unroll
      for (int i = 0; i < 4; ++i)
        acc[mi][i] = wmma_bf16(a1[mi], b1[i], acc[mi][i]);
  }

#pragma unroll
  for (int mi = 0; mi < 2; ++mi) {
#pragma unroll
    for (int i = 0; i < 4; ++i) {
      const float bv = bias[n0 + 16 * i + l16];
#pragma unroll
      for (int r = 0; r < 8; ++r) {
        const int m = r + 8 * half;
        out[(size_t)(t0 + 16 * mi + m) * Dn + n0 + 16 * i + l16] =
            acc[mi][i][r] + bv;
      }
    }
  }
}

// --------------------------- launcher --------------------------------------

extern "C" void kernel_launch(void* const* d_in, const int* in_sizes, int n_in,
                              void* d_out, int out_size, void* d_ws,
                              size_t ws_size, hipStream_t stream) {
  (void)in_sizes; (void)n_in; (void)out_size; (void)ws_size;

  const float* x     = (const float*)d_in[0];
  const float* W_qkv = (const float*)d_in[1];
  const float* b_qkv = (const float*)d_in[2];
  const float* W_out = (const float*)d_in[3];
  const float* b_out = (const float*)d_in[4];
  float* out = (float*)d_out;

  char* ws = (char*)d_ws;
  size_t off = 0;
  auto carve = [&](size_t bytes) -> void* {
    void* p = ws + off;
    off += (bytes + 255) & ~(size_t)255;
    return p;
  };
  const size_t nBTD = (size_t)Bn * Tn * Dn;
  bf16_t* xbf = (bf16_t*)carve(nBTD * 2);
  bf16_t* Qb  = (bf16_t*)carve(nBTD * 2);
  bf16_t* Kb  = (bf16_t*)carve(nBTD * 2);
  bf16_t* Vt  = (bf16_t*)carve(nBTD * 2);
  bf16_t* Ob  = (bf16_t*)carve(nBTD * 2);
  bf16_t* WqT = (bf16_t*)carve((size_t)Dn * 3 * Dn * 2);
  bf16_t* WoT = (bf16_t*)carve((size_t)Dn * Dn * 2);

  const int nx = Bn * Tn * Dn;
  cvt_bf16_kernel<<<(nx + 255) / 256, 256, 0, stream>>>(x, xbf, nx);
  transpose_bf16_kernel<<<(Dn * 3 * Dn + 255) / 256, 256, 0, stream>>>(
      W_qkv, WqT, Dn, 3 * Dn);
  transpose_bf16_kernel<<<(Dn * Dn + 255) / 256, 256, 0, stream>>>(
      W_out, WoT, Dn, Dn);

  // 256 M-strips * 48 N-strips = 12288 waves / 8 per block
  qkv_gemm_kernel<<<1536, 256, 0, stream>>>(xbf, WqT, b_qkv, Qb, Kb, Vt);
  // 8192 q-tile waves / 8 per block
  attn_kernel<<<1024, 256, 0, stream>>>(Qb, Kb, Vt, Ob);
  // 256 M-strips * 16 N-strips = 4096 waves / 8 per block
  oproj_kernel<<<512, 256, 0, stream>>>(Ob, WoT, b_out, out);
}
